// NoisyTopKRouter_76974403879709
// MI455X (gfx1250) — matrix operand order repaired and checked by reading, not compiled, not run on hardware
//
#include <hip/hip_runtime.h>

#define N_ROWS 16384
#define DIM    4096
#define NEXP   64
#define TOPK   8
#define WAVES  4
#define ROWS_PER_WAVE 16
#define ROWS_PER_WG   (WAVES * ROWS_PER_WAVE)

typedef __bf16 bf16;
typedef __bf16 v16bf __attribute__((ext_vector_type(16)));
typedef float  v8f   __attribute__((ext_vector_type(8)));

// ---------------------------------------------------------------------------
// Prep: transpose + convert Wg/Wn [K=4096, E=64] fp32 -> [E][K] bf16 in d_ws.
// Makes the WMMA B operand (per-expert column, K-contiguous) two contiguous
// 16B loads per lane. 512 KB per matrix; lives in L2 for the whole launch.
// ---------------------------------------------------------------------------
__global__ __launch_bounds__(256) void prep_weights(const float* __restrict__ Wg,
                                                    const float* __restrict__ Wn,
                                                    bf16* __restrict__ wt) {
    int idx = blockIdx.x * blockDim.x + threadIdx.x;   // over NEXP*DIM
    int e = idx / DIM;
    int k = idx - e * DIM;
    wt[idx]                        = (bf16)Wg[(size_t)k * NEXP + e];
    wt[(size_t)NEXP * DIM + idx]   = (bf16)Wn[(size_t)k * NEXP + e];
}

// Build A operand (16x32 bf16 tile):
// lane L / L+16 hold row (L&15); K elements {koff..koff+15} -> a[0..15],
// koff = (lane>>4)*16.
__device__ __forceinline__ v16bf load_a_bf16(const float* __restrict__ p) {
    float4 f0 = *reinterpret_cast<const float4*>(p + 0);
    float4 f1 = *reinterpret_cast<const float4*>(p + 4);
    float4 f2 = *reinterpret_cast<const float4*>(p + 8);
    float4 f3 = *reinterpret_cast<const float4*>(p + 12);
    v16bf a;
    a[0]  = (bf16)f0.x; a[1]  = (bf16)f0.y; a[2]  = (bf16)f0.z; a[3]  = (bf16)f0.w;
    a[4]  = (bf16)f1.x; a[5]  = (bf16)f1.y; a[6]  = (bf16)f1.z; a[7]  = (bf16)f1.w;
    a[8]  = (bf16)f2.x; a[9]  = (bf16)f2.y; a[10] = (bf16)f2.z; a[11] = (bf16)f2.w;
    a[12] = (bf16)f3.x; a[13] = (bf16)f3.y; a[14] = (bf16)f3.z; a[15] = (bf16)f3.w;
    return a;
}

__device__ __forceinline__ void load_stage(const float* __restrict__ xl,
                                           const bf16* const* bgp,
                                           const bf16* const* bnp,
                                           int kc, v16bf& a, v16bf* g, v16bf* n) {
    a = load_a_bf16(xl + kc);
#pragma unroll
    for (int t = 0; t < 4; ++t) {
        g[t] = *reinterpret_cast<const v16bf*>(bgp[t] + kc);
        n[t] = *reinterpret_cast<const v16bf*>(bnp[t] + kc);
    }
}

__device__ __forceinline__ void wmma8(const v16bf& a, const v16bf* g, const v16bf* n,
                                      v8f* accG, v8f* accN) {
#pragma unroll
    for (int t = 0; t < 4; ++t) {
        accG[t] = __builtin_amdgcn_wmma_f32_16x16x32_bf16(
            false, a, false, g[t], (short)0, accG[t], false, false);
        accN[t] = __builtin_amdgcn_wmma_f32_16x16x32_bf16(
            false, a, false, n[t], (short)0, accN[t], false, false);
    }
}

__global__ __launch_bounds__(WAVES * 32) void noisy_topk_router(
    const float* __restrict__ x,
    const bf16*  __restrict__ wtg,   // [E][K] bf16
    const bf16*  __restrict__ wtn,   // [E][K] bf16
    const float* __restrict__ bg,
    const float* __restrict__ bn,
    const float* __restrict__ eps,
    float* __restrict__ gates,
    float* __restrict__ logits) {

    __shared__ float lbuf[WAVES][ROWS_PER_WAVE][NEXP];   // noisy logits (row-major)
    __shared__ float gbuf[WAVES][ROWS_PER_WAVE][NEXP];   // gates rows

    const int tid  = threadIdx.x;
    const int w    = tid >> 5;          // wave in block
    const int lane = tid & 31;
    const int ln   = lane & 15;
    const int lg   = lane >> 4;         // lane group 0/1
    const int row0 = (blockIdx.x * WAVES + w) * ROWS_PER_WAVE;

    // Per-lane base pointers
    const float* xl = x + (size_t)(row0 + ln) * DIM + lg * 16;      // A operand
    const bf16* bgp[4];
    const bf16* bnp[4];
#pragma unroll
    for (int t = 0; t < 4; ++t) {
        const int col = t * 16 + ln;                                 // expert column
        bgp[t] = wtg + (size_t)col * DIM + lg * 16;                  // B operand: 16 contig K
        bnp[t] = wtn + (size_t)col * DIM + lg * 16;
    }

    v8f accG[4] = {{0.f}};
    v8f accN[4] = {{0.f}};

    // --- 2x-unrolled ping-pong pipeline: 128 stages of K=32, 8 WMMAs/stage.
    // Ping-pong register sets (buf0/buf1) make the double-buffer hand-off a
    // compile-time rename: no v_mov copies, no WMMA->VALU WAR hazard NOPs.
    v16bf a0, a1;
    v16bf g0[4], n0[4], g1[4], n1[4];

    load_stage(xl, bgp, bnp, 0, a0, g0, n0);            // stage 0 -> buf0

    int kc = 32;
#pragma unroll 1
    for (int it = 0; it < 63; ++it) {
        load_stage(xl, bgp, bnp, kc, a1, g1, n1);       // stage kc   -> buf1
        wmma8(a0, g0, n0, accG, accN);                  // consume stage kc-32
        load_stage(xl, bgp, bnp, kc + 32, a0, g0, n0);  // stage kc+32 -> buf0
        wmma8(a1, g1, n1, accG, accN);                  // consume stage kc
        kc += 64;
    }
    // kc == 4064 here; buf0 holds stage 4032.
    load_stage(xl, bgp, bnp, kc, a1, g1, n1);           // last stage 4064 -> buf1
    wmma8(a0, g0, n0, accG, accN);                      // stage 4032
    wmma8(a1, g1, n1, accG, accN);                      // stage 4064

    // --- epilogue: bias, softplus noise, noisy logits -> global + LDS ---
    // C/D layout: reg v, lanes 0-15 -> row v, lanes 16-31 -> row v+8; N = lane&15.
#pragma unroll
    for (int t = 0; t < 4; ++t) {
        const int e = t * 16 + ln;
        const float bgv = bg[e];
        const float bnv = bn[e];
#pragma unroll
        for (int v = 0; v < 8; ++v) {
            const int r    = v + 8 * lg;
            const int grow = row0 + r;
            const float gl = accG[t][v] + bgv;
            const float nz = accN[t][v] + bnv;
            const float sp = fmaxf(nz, 0.f) + log1pf(__expf(-fabsf(nz)));  // softplus
            const float lt = gl + eps[(size_t)grow * NEXP + e] * (sp + 0.01f);
            logits[(size_t)grow * NEXP + e] = lt;
            lbuf[w][r][e] = lt;
        }
    }

    asm volatile("s_wait_dscnt 0" ::: "memory");

    // --- per-row top-8 + softmax + scatter into LDS gates row ---
    if (lane < 16) {
        const float* rowp  = &lbuf[w][lane][0];
        float*       growp = &gbuf[w][lane][0];
        unsigned long long mask = 0ull;
        float topv[TOPK];
        int   topi[TOPK];
#pragma unroll 1
        for (int k = 0; k < TOPK; ++k) {
            float best = -3.4e38f;
            int   bi   = 0;
#pragma unroll 1
            for (int e = 0; e < NEXP; ++e) {
                const bool taken = (mask >> e) & 1ull;
                const float vv = rowp[e];
                if (!taken && vv > best) { best = vv; bi = e; }
            }
            mask |= (1ull << bi);
            topv[k] = best;
            topi[k] = bi;
        }
        const float m = topv[0];   // first pass found the max
        float p[TOPK];
        float s = 0.f;
#pragma unroll
        for (int k = 0; k < TOPK; ++k) { p[k] = __expf(topv[k] - m); s += p[k]; }
        const float inv = 1.0f / s;
#pragma unroll 1
        for (int e = 0; e < NEXP; ++e) growp[e] = 0.f;
#pragma unroll
        for (int k = 0; k < TOPK; ++k) growp[topi[k]] = p[k] * inv;
    }

    asm volatile("s_wait_dscnt 0" ::: "memory");

    // --- coalesced cooperative store of gates ---
#pragma unroll 1
    for (int i = lane; i < ROWS_PER_WAVE * NEXP; i += 32) {
        const int r = i >> 6;
        const int e = i & 63;
        gates[(size_t)(row0 + r) * NEXP + e] = gbuf[w][r][e];
    }
}

extern "C" void kernel_launch(void* const* d_in, const int* in_sizes, int n_in,
                              void* d_out, int out_size, void* d_ws, size_t ws_size,
                              hipStream_t stream) {
    const float* x   = (const float*)d_in[0];
    const float* Wg  = (const float*)d_in[1];
    const float* bg  = (const float*)d_in[2];
    const float* Wn  = (const float*)d_in[3];
    const float* bn  = (const float*)d_in[4];
    const float* eps = (const float*)d_in[5];

    float* gates  = (float*)d_out;                          // [N, E]
    float* logits = (float*)d_out + (size_t)N_ROWS * NEXP;  // [N, E]

    bf16* wt = (bf16*)d_ws;   // [2][E][K] bf16 = 1 MB

    prep_weights<<<(NEXP * DIM) / 256, 256, 0, stream>>>(Wg, Wn, wt);

    noisy_topk_router<<<N_ROWS / ROWS_PER_WG, WAVES * 32, 0, stream>>>(
        x, wt, wt + (size_t)NEXP * DIM, bg, bn, eps, gates, logits);
}
